// DeepSSM_80582176408251
// MI455X (gfx1250) — matrix operand order repaired and checked by reading, not compiled, not run on hardware
//
#include <hip/hip_runtime.h>

// LSTM: B=256, T=2048, OBS=32, HID=64, gates 4H=256.
// 16 blocks x 128 threads (4 waves). Wave w owns hidden cols [16w,16w+16).
// Weights register-resident as f16 WMMA B-fragments; h/x double-buffered in LDS.
// Elementwise: hardware v_rcp_f32 + (if available) hardware v_tanh_f32.

#define OBS   32
#define HID   64
#define G4H   256   // 4*HID
#define TSTEPS 2048
#define BATCH 256

typedef __attribute__((ext_vector_type(16))) _Float16 v16h;
typedef __attribute__((ext_vector_type(8)))  _Float16 v8h;
typedef __attribute__((ext_vector_type(8)))  float    v8f;

__device__ __forceinline__ float fast_rcp(float x) {
    return __builtin_amdgcn_rcpf(x);   // v_rcp_f32, ~1 ulp
}
__device__ __forceinline__ float fast_sigmoid(float x) {
    return fast_rcp(1.0f + __expf(-x));
}
__device__ __forceinline__ float fast_tanh(float x) {
#if __has_builtin(__builtin_amdgcn_tanhf)
    return __builtin_amdgcn_tanhf(x);  // native v_tanh_f32 (gfx1250 TRANS op)
#else
    float ax = fabsf(x);
    float e  = __expf(-2.0f * ax);
    float t  = (1.0f - e) * fast_rcp(1.0f + e);
    return copysignf(t, x);
#endif
}

// Gather a 16x32 f16 A-fragment row for this lane from an LDS row pointer.
// A k order chosen here (contiguous, same as B): lanes 0-15: elems 0-15 = K0..15;
// lanes 16-31: elems 0-15 = K16..31.  => two b128 loads.
__device__ __forceinline__ v16h ldA(const _Float16* p, int halfsel) {
    v8h lo = *(const v8h*)(p + halfsel * 16);
    v8h hi = *(const v8h*)(p + halfsel * 16 + 8);
    v16h r;
#pragma unroll
    for (int i = 0; i < 8; ++i) { r[i] = lo[i]; r[8 + i] = hi[i]; }
    return r;
}

__global__ __launch_bounds__(128, 1)
void lstm_wmma_kernel(const float* __restrict__ y,      // [B, T, OBS]
                      const float* __restrict__ Wx,     // [OBS, 4H]
                      const float* __restrict__ Wh,     // [HID, 4H]
                      const float* __restrict__ bias_g, // [4H]
                      float* __restrict__ out)          // [B, T, HID]
{
    __shared__ __align__(16) _Float16 xbuf[2][16][OBS];  // 2 KB
    __shared__ __align__(16) _Float16 hbuf[2][16][HID];  // 4 KB

    const int tid     = threadIdx.x;
    const int wave    = tid >> 5;        // 0..3 : hidden col slice
    const int lane    = tid & 31;
    const int halfsel = lane >> 4;       // 0/1
    const int ncol    = lane & 15;       // column within 16-wide tile
    const int rowM    = ncol;            // A-matrix row for this lane
    const int tileB   = blockIdx.x * 16; // batch rows [tileB, tileB+16)
    const int hcol    = wave * 16 + ncol;

    // ---- one-time: load register-resident weight (B) fragments, f16 ----
    // B k order chosen here (contiguous, same as ldA): lane half 0 holds K base..base+15 (packed pairs),
    // lane half 1 holds K base+16..base+31; N = lane&15.
    v16h bx[4];        // Wx tile per gate, K=32
    v16h bh[4][2];     // Wh tiles per gate, K chunks 0..31 / 32..63
    float bval[4];
#pragma unroll
    for (int g = 0; g < 4; ++g) {
        const int col = g * HID + hcol;
        v16h t;
#pragma unroll
        for (int j = 0; j < 16; ++j) {
            const int k = halfsel * 16 + j;
            t[j] = (_Float16)Wx[k * G4H + col];
        }
        bx[g] = t;
#pragma unroll
        for (int c = 0; c < 2; ++c) {
            v16h u;
#pragma unroll
            for (int j = 0; j < 16; ++j) {
                const int k = c * 32 + halfsel * 16 + j;
                u[j] = (_Float16)Wh[k * G4H + col];
            }
            bh[g][c] = u;
        }
        bval[g] = bias_g[col];
    }

    // ---- init: h0 = 0 in LDS, stage x_0, c0 = 0 in regs ----
    for (int i = tid; i < 16 * HID; i += 128)
        ((_Float16*)hbuf[0])[i] = (_Float16)0.0f;
    {
        const int m = tid >> 3, grp = tid & 7;
        const float4 v = *(const float4*)(y + ((size_t)(tileB + m) * TSTEPS) * OBS + grp * 4);
        _Float16* dst = &xbuf[0][m][grp * 4];
        dst[0] = (_Float16)v.x; dst[1] = (_Float16)v.y;
        dst[2] = (_Float16)v.z; dst[3] = (_Float16)v.w;
    }
    float cst[8];
#pragma unroll
    for (int r = 0; r < 8; ++r) cst[r] = 0.0f;
    __syncthreads();

    // ---- recurrence ----
    for (int t = 0; t < TSTEPS; ++t) {
        const int cur = t & 1, nxt = cur ^ 1;

        // A fragments from LDS (ds_load_b128 x6)
        const v16h ax  = ldA(&xbuf[cur][rowM][0],  halfsel);
        const v16h ah0 = ldA(&hbuf[cur][rowM][0],  halfsel);
        const v16h ah1 = ldA(&hbuf[cur][rowM][32], halfsel);

        // gates = x@Wx + h@Wh  (f16 WMMA, f32 accumulate), 12 WMMAs/wave/step
        v8f acc[4];
#pragma unroll
        for (int g = 0; g < 4; ++g) {
            v8f a = {};
            a = __builtin_amdgcn_wmma_f32_16x16x32_f16(false, ax,  false, bx[g],    (short)0, a, false, false);
            a = __builtin_amdgcn_wmma_f32_16x16x32_f16(false, ah0, false, bh[g][0], (short)0, a, false, false);
            a = __builtin_amdgcn_wmma_f32_16x16x32_f16(false, ah1, false, bh[g][1], (short)0, a, false, false);
            acc[g] = a;
        }

        // prefetch x_{t+1} into the other LDS buffer while we do elementwise
        if (t + 1 < TSTEPS) {
            const int m = tid >> 3, grp = tid & 7;
            const float4 v = *(const float4*)(y + ((size_t)(tileB + m) * TSTEPS + (t + 1)) * OBS + grp * 4);
            _Float16* dst = &xbuf[nxt][m][grp * 4];
            dst[0] = (_Float16)v.x; dst[1] = (_Float16)v.y;
            dst[2] = (_Float16)v.z; dst[3] = (_Float16)v.w;
        }

        // elementwise LSTM cell update; C/D layout: vgpr r -> M = r + 8*halfsel, N = ncol
#pragma unroll
        for (int r = 0; r < 8; ++r) {
            const int M  = r + halfsel * 8;
            const float iv = fast_sigmoid(acc[0][r] + bval[0]);
            const float fv = fast_sigmoid(acc[1][r] + bval[1]);
            const float gv = fast_tanh   (acc[2][r] + bval[2]);
            const float ov = fast_sigmoid(acc[3][r] + bval[3]);
            const float cn = fv * cst[r] + iv * gv;
            cst[r] = cn;
            const float hn = ov * fast_tanh(cn);
            hbuf[nxt][M][hcol] = (_Float16)hn;
            out[(((size_t)(tileB + M)) * TSTEPS + t) * HID + hcol] = hn;
        }
        __syncthreads();  // h_{t+1} and x_{t+1} visible; buffers safe to swap
    }
}

extern "C" void kernel_launch(void* const* d_in, const int* in_sizes, int n_in,
                              void* d_out, int out_size, void* d_ws, size_t ws_size,
                              hipStream_t stream) {
    const float* y  = (const float*)d_in[0];
    const float* Wx = (const float*)d_in[1];
    const float* Wh = (const float*)d_in[2];
    const float* b  = (const float*)d_in[3];
    float* out = (float*)d_out;
    lstm_wmma_kernel<<<dim3(BATCH / 16), dim3(128), 0, stream>>>(y, Wx, Wh, b, out);
}
